// MultiHeadAttention_56418690400728
// MI455X (gfx1250) — matrix operand order repaired and checked by reading, not compiled, not run on hardware
//
#include <hip/hip_runtime.h>
#include <hip/hip_bf16.h>

// ---------------------------------------------------------------------------
// MHA forward for MI455X (gfx1250, wave32, WMMA + async-to-LDS DMA).
// f32->f16 convert (weights transposed) -> 3x projection GEMM (WMMA, double-
// buffered async LDS staging; Q/K head-split, V head-split-transposed) ->
// flash attention (WMMA QK^T + PV, online softmax, async K/V staging) ->
// output GEMM (WMMA, f32 + bias into d_out).
// Workspace: q/k/v f16 (48MB) + wT f16 (8MB) + Qh/Kh/VhT (48MB) + AO (16MB).
// ---------------------------------------------------------------------------

typedef _Float16 v16h __attribute__((ext_vector_type(16)));
typedef float    v8f  __attribute__((ext_vector_type(8)));

#define D_MODEL  1024
#define N_HEADS  16
#define HEAD_DIM 64
#define SEQ      2048
#define BATCH    4
#define NTOK     (BATCH * SEQ)   // 8192

__device__ __forceinline__ v8f vzero() {
  v8f z;
#pragma unroll
  for (int i = 0; i < 8; ++i) z[i] = 0.0f;
  return z;
}

__device__ __forceinline__ v8f wmma_f16(v16h a, v16h b, v8f c) {
  return __builtin_amdgcn_wmma_f32_16x16x32_f16(false, a, false, b, (short)0, c,
                                                false, false);
}

// Async DMA: global -> LDS, 16B per lane, tracked by ASYNCcnt (no VGPR data).
__device__ __forceinline__ void async_copy_b128(void* lds, const void* g) {
  asm volatile("global_load_async_to_lds_b128 %0, %1, off"
               :: "v"((unsigned)(unsigned long long)lds), "v"(g)
               : "memory");
}
__device__ __forceinline__ void wait_async() {
  asm volatile("s_wait_asynccnt 0" ::: "memory");
}

// A-fragment (16x32 f16, MxK) from LDS [outer][k] row-major.
// lane<16: row=lane, K={0..7,16..23}; lane>=16: row=lane-16, K={8..15,24..31}.
__device__ __forceinline__ v16h load_frag_a(const _Float16* p, int stride,
                                            int outer, int koff) {
  const int lane = threadIdx.x & 31;
  const int half = lane >> 4, l16 = lane & 15;
  const _Float16* q = p + (outer + l16) * stride + koff + 8 * half;
  v16h a;
#pragma unroll
  for (int e = 0; e < 8; ++e) a[e] = q[e];
#pragma unroll
  for (int e = 0; e < 8; ++e) a[8 + e] = q[16 + e];
  return a;
}

// B-fragment (32x16 f16, KxN) from LDS [col][k] col-major.
// lanes 0-15: K={0..7,16..23}, lanes 16-31: K={8..15,24..31}, col = lane&15.
__device__ __forceinline__ v16h load_frag_b(const _Float16* p, int stride,
                                            int outer, int koff) {
  const int lane = threadIdx.x & 31;
  const int half = lane >> 4, l16 = lane & 15;
  const _Float16* q = p + (outer + l16) * stride + koff + 8 * half;
  v16h b;
#pragma unroll
  for (int e = 0; e < 8; ++e) b[e] = q[e];
#pragma unroll
  for (int e = 0; e < 8; ++e) b[8 + e] = q[16 + e];
  return b;
}

// ---------------------------------------------------------------------------
// f32 -> f16, 8 elems/thread, 16B vector in/out.
// ---------------------------------------------------------------------------
__global__ __launch_bounds__(256) void cvt_f32_f16(const float* __restrict__ in,
                                                   _Float16* __restrict__ out,
                                                   int n8) {
  int i = blockIdx.x * blockDim.x + threadIdx.x;
  if (i >= n8) return;
  const float4* p = (const float4*)in + (size_t)i * 2;
  float4 x = p[0], y = p[1];
  _Float16 t[8] = {(_Float16)x.x, (_Float16)x.y, (_Float16)x.z, (_Float16)x.w,
                   (_Float16)y.x, (_Float16)y.y, (_Float16)y.z, (_Float16)y.w};
  *(uint4*)(out + (size_t)i * 8) = *(uint4*)t;
}

// f32 [k][n] -> f16 [n][k] (64x64 LDS tile transpose). grid (16,16).
__global__ __launch_bounds__(256) void cvt_transpose_f16(
    const float* __restrict__ in, _Float16* __restrict__ out) {
  __shared__ _Float16 t[64][65];
  const int tid = threadIdx.x;
  const int bn = blockIdx.x * 64, bk = blockIdx.y * 64;
#pragma unroll
  for (int i = 0; i < 16; ++i) {
    int c = tid + i * 256;              // 0..4095
    int k = c >> 6, n = c & 63;
    t[n][k] = (_Float16)in[(size_t)(bk + k) * D_MODEL + bn + n];
  }
  __syncthreads();
#pragma unroll
  for (int i = 0; i < 16; ++i) {
    int c = tid + i * 256;
    int n = c >> 6, k = c & 63;
    out[(size_t)(bn + n) * D_MODEL + bk + k] = t[n][k];
  }
}

// ---------------------------------------------------------------------------
// WMMA GEMM: C[8192,1024] = A[8192,1024] * W + bias, W given transposed [n][k].
// 128x128 block, K-step 32, 256 thr = 8 waves (4x2), wave tile 32x64.
// Double-buffered LDS, all staging via async-to-LDS DMA.
// mode 0: f16 head-split [B,H,S,HD]; mode 2: f16 [B,H,HD,S]; mode 1: f32 flat.
// ---------------------------------------------------------------------------
#define GT_M 128
#define GT_N 128
#define GT_K 32
#define LDT  40   // padded f16 stride (rows 80B: 16B-aligned, conflict-spread)

__global__ __launch_bounds__(256) void gemm_wmma(
    const _Float16* __restrict__ A, const _Float16* __restrict__ WT,
    const float* __restrict__ bias, _Float16* __restrict__ outHS,
    float* __restrict__ outF32, int mode) {
  __shared__ _Float16 As[2][GT_M * LDT];  // [row][k]
  __shared__ _Float16 Bs[2][GT_N * LDT];  // [col][k]

  const int tid  = threadIdx.x;
  const int lane = tid & 31;
  const int wave = tid >> 5;
  const int wm   = wave >> 1;  // 0..3 : 32-row slice
  const int wn   = wave & 1;   // 0..1 : 64-col slice
  const int half = lane >> 4, l16 = lane & 15;

  const int gm = blockIdx.x * GT_M;
  const int gn = blockIdx.y * GT_N;

  auto stage = [&](int buf, int kt) {
#pragma unroll
    for (int i = 0; i < 2; ++i) {
      int c = tid + i * 256;            // 0..511
      int row = c >> 2;
      int kc = (c & 3) * 8;
      async_copy_b128(&As[buf][row * LDT + kc],
                      A + (size_t)(gm + row) * D_MODEL + kt + kc);
      async_copy_b128(&Bs[buf][row * LDT + kc],
                      WT + (size_t)(gn + row) * D_MODEL + kt + kc);
    }
  };

  v8f acc[2][4];
#pragma unroll
  for (int mi = 0; mi < 2; ++mi)
#pragma unroll
    for (int ni = 0; ni < 4; ++ni) acc[mi][ni] = vzero();

  stage(0, 0);
  int buf = 0;
  for (int kt = 0; kt < D_MODEL; kt += GT_K) {
    wait_async();      // my DMA done ...
    __syncthreads();   // ... and everyone's (tile `buf` now valid for all)
    if (kt + GT_K < D_MODEL) stage(buf ^ 1, kt + GT_K);  // overlap next tile

    v16h af[2];
#pragma unroll
    for (int mi = 0; mi < 2; ++mi)
      af[mi] = load_frag_a(As[buf], LDT, wm * 32 + mi * 16, 0);
#pragma unroll
    for (int ni = 0; ni < 4; ++ni) {
      v16h bf = load_frag_b(Bs[buf], LDT, wn * 64 + ni * 16, 0);
#pragma unroll
      for (int mi = 0; mi < 2; ++mi)
        acc[mi][ni] = wmma_f16(af[mi], bf, acc[mi][ni]);
    }
    buf ^= 1;
  }

#pragma unroll
  for (int mi = 0; mi < 2; ++mi) {
#pragma unroll
    for (int ni = 0; ni < 4; ++ni) {
      int n = gn + wn * 64 + ni * 16 + l16;
      float bn = bias[n];
#pragma unroll
      for (int r = 0; r < 8; ++r) {
        int m = gm + wm * 32 + mi * 16 + r + 8 * half;
        float val = acc[mi][ni][r] + bn;
        if (mode == 1) {
          outF32[(size_t)m * D_MODEL + n] = val;
        } else {
          int b = m >> 11, s = m & (SEQ - 1);
          int h = n >> 6, hd = n & 63;
          size_t idx = (mode == 0)
                           ? (((size_t)b * N_HEADS + h) * SEQ + s) * HEAD_DIM + hd
                           : (((size_t)b * N_HEADS + h) * HEAD_DIM + hd) * SEQ + s;
          outHS[idx] = (_Float16)val;
        }
      }
    }
  }
}

// ---------------------------------------------------------------------------
// Flash attention: workgroup = (b, h, 64 q-rows); 4 waves x 16 q-rows.
// 64-key blocks: QK^T (WMMA) -> online softmax (LDS) -> PV (WMMA).
// K staged [key][hd], V staged [hd][key] from the pre-transposed VhT — all
// tiles arrive via async-to-LDS DMA, all fragment reads are contiguous.
// ---------------------------------------------------------------------------
#define FA_QB  64
#define FA_KB  64
#define FA_LD  72   // f16 stride (144B rows: 16B-aligned)
#define FA_SLD 66   // f32 score stride (bank-conflict padding)

__global__ __launch_bounds__(128) void flash_attn(
    const _Float16* __restrict__ Qh, const _Float16* __restrict__ Kh,
    const _Float16* __restrict__ VhT, _Float16* __restrict__ O) {
  __shared__ _Float16 Qs[FA_QB * FA_LD];     // [qrow][hd]
  __shared__ _Float16 Ks[FA_KB * FA_LD];     // [key][hd]  (B col = key)
  __shared__ _Float16 Vt[HEAD_DIM * FA_LD];  // [hd][key]  (B col = hd)
  __shared__ _Float16 Ps[4][16 * FA_LD];     // per-wave P tile [row][key]
  __shared__ float    Ss[4][16 * FA_SLD];    // per-wave raw scores
  __shared__ float    bc[4][16];             // per-wave row broadcast

  const int tid  = threadIdx.x;
  const int lane = tid & 31;
  const int wave = tid >> 5;
  const int half = lane >> 4, l16 = lane & 15;

  const int bh = blockIdx.y;
  const int b  = bh >> 4;
  const int h  = bh & 15;
  const int qbase = blockIdx.x * FA_QB;
  const size_t hp = (size_t)bh * SEQ * HEAD_DIM;  // same for [S,HD] and [HD,S]

  // Q block: issue DMA now; first in-loop wait covers it.
#pragma unroll
  for (int i = 0; i < 4; ++i) {
    int c = tid + i * 128;              // 0..511
    int row = c >> 3, k8 = (c & 7) * 8;
    async_copy_b128(&Qs[row * FA_LD + k8],
                    Qh + hp + (size_t)(qbase + row) * HEAD_DIM + k8);
  }

  v8f oacc[4];
#pragma unroll
  for (int ni = 0; ni < 4; ++ni) oacc[ni] = vzero();
  float m_i = -3.0e38f, l_i = 0.0f;
  const int srow = l16;              // softmax row (halves redundant)
  const float sscale = 0.125f;       // 1/sqrt(64), applied in f32

  for (int kb = 0; kb < SEQ; kb += FA_KB) {
    __syncthreads();   // prior reads of Ks/Vt/Ps/bc complete
#pragma unroll
    for (int i = 0; i < 4; ++i) {
      int c = tid + i * 128;
      int row = c >> 3, k8 = (c & 7) * 8;   // row = key for K, = hd for V
      async_copy_b128(&Ks[row * FA_LD + k8],
                      Kh + hp + (size_t)(kb + row) * HEAD_DIM + k8);
      async_copy_b128(&Vt[row * FA_LD + k8],
                      VhT + hp + (size_t)row * SEQ + kb + k8);
    }
    wait_async();
    __syncthreads();

    // S = Q K^T : 16 q-rows x 64 keys (K-dim = HEAD_DIM, 2 WMMA k-steps).
    v16h aq0 = load_frag_a(Qs, FA_LD, wave * 16, 0);
    v16h aq1 = load_frag_a(Qs, FA_LD, wave * 16, 32);
#pragma unroll
    for (int ni = 0; ni < 4; ++ni) {
      v8f st = vzero();
      st = wmma_f16(aq0, load_frag_b(Ks, FA_LD, ni * 16, 0), st);
      st = wmma_f16(aq1, load_frag_b(Ks, FA_LD, ni * 16, 32), st);
#pragma unroll
      for (int r = 0; r < 8; ++r)
        Ss[wave][(r + 8 * half) * FA_SLD + ni * 16 + l16] = st[r];
    }
    __syncthreads();

    // Online softmax: each lane owns one score row of 64.
    const float* sp = &Ss[wave][srow * FA_SLD];
    float rmax = -3.0e38f;
#pragma unroll
    for (int j = 0; j < FA_KB; ++j) rmax = fmaxf(rmax, sp[j]);
    float mnew = fmaxf(m_i, rmax * sscale);
    float alpha = __expf(m_i - mnew);
    float rsum = 0.0f;
    _Float16* pp = &Ps[wave][srow * FA_LD];
#pragma unroll
    for (int j = 0; j < FA_KB; ++j) {
      float p = __expf(sp[j] * sscale - mnew);
      rsum += p;
      pp[j] = (_Float16)p;
    }
    m_i = mnew;
    l_i = l_i * alpha + rsum;
    if (half == 0) bc[wave][srow] = alpha;
    __syncthreads();

    // Rescale O, accumulate P @ V.
    float al[8];
#pragma unroll
    for (int r = 0; r < 8; ++r) al[r] = bc[wave][r + 8 * half];
    v16h ap0 = load_frag_a(Ps[wave], FA_LD, 0, 0);
    v16h ap1 = load_frag_a(Ps[wave], FA_LD, 0, 32);
#pragma unroll
    for (int ni = 0; ni < 4; ++ni) {
#pragma unroll
      for (int r = 0; r < 8; ++r) oacc[ni][r] *= al[r];
      oacc[ni] = wmma_f16(ap0, load_frag_b(Vt, FA_LD, ni * 16, 0), oacc[ni]);
      oacc[ni] = wmma_f16(ap1, load_frag_b(Vt, FA_LD, ni * 16, 32), oacc[ni]);
    }
  }

  // Normalize and write merged-head f16 output.
  if (half == 0) bc[wave][srow] = 1.0f / l_i;
  __syncthreads();
  float sc[8];
#pragma unroll
  for (int r = 0; r < 8; ++r) sc[r] = bc[wave][r + 8 * half];
#pragma unroll
  for (int ni = 0; ni < 4; ++ni) {
#pragma unroll
    for (int r = 0; r < 8; ++r) {
      int qrow = qbase + wave * 16 + r + 8 * half;
      int col = h * HEAD_DIM + ni * 16 + l16;
      O[(size_t)(b * SEQ + qrow) * D_MODEL + col] =
          (_Float16)(oacc[ni][r] * sc[r]);
    }
  }
}

// ---------------------------------------------------------------------------
extern "C" void kernel_launch(void* const* d_in, const int* in_sizes, int n_in,
                              void* d_out, int out_size, void* d_ws,
                              size_t ws_size, hipStream_t stream) {
  (void)in_sizes; (void)n_in; (void)out_size; (void)ws_size;
  const float* q  = (const float*)d_in[0];
  const float* k  = (const float*)d_in[1];
  const float* v  = (const float*)d_in[2];
  const float* wq = (const float*)d_in[3];
  const float* bq = (const float*)d_in[4];
  const float* wk = (const float*)d_in[5];
  const float* bk = (const float*)d_in[6];
  const float* wv = (const float*)d_in[7];
  const float* bv = (const float*)d_in[8];
  const float* wo = (const float*)d_in[9];
  const float* bo = (const float*)d_in[10];

  const size_t TOK = (size_t)NTOK * D_MODEL;     // 8388608 elems
  const size_t WEL = (size_t)D_MODEL * D_MODEL;  // 1048576 elems
  _Float16* ws   = (_Float16*)d_ws;
  _Float16* q16  = ws; ws += TOK;
  _Float16* k16  = ws; ws += TOK;
  _Float16* vx16 = ws; ws += TOK;
  _Float16* wqT  = ws; ws += WEL;
  _Float16* wkT  = ws; ws += WEL;
  _Float16* wvT  = ws; ws += WEL;
  _Float16* woT  = ws; ws += WEL;
  _Float16* Qh   = ws; ws += TOK;
  _Float16* Kh   = ws; ws += TOK;
  _Float16* VhT  = ws; ws += TOK;
  _Float16* AO   = ws; ws += TOK;

  const int t8 = (int)(TOK / 8);
  cvt_f32_f16<<<(t8 + 255) / 256, 256, 0, stream>>>(q, q16, t8);
  cvt_f32_f16<<<(t8 + 255) / 256, 256, 0, stream>>>(k, k16, t8);
  cvt_f32_f16<<<(t8 + 255) / 256, 256, 0, stream>>>(v, vx16, t8);
  dim3 tg(D_MODEL / 64, D_MODEL / 64);
  cvt_transpose_f16<<<tg, 256, 0, stream>>>(wq, wqT);
  cvt_transpose_f16<<<tg, 256, 0, stream>>>(wk, wkT);
  cvt_transpose_f16<<<tg, 256, 0, stream>>>(wv, wvT);
  cvt_transpose_f16<<<tg, 256, 0, stream>>>(wo, woT);

  dim3 gg(NTOK / GT_M, D_MODEL / GT_N);
  gemm_wmma<<<gg, 256, 0, stream>>>(q16, wqT, bq, Qh, nullptr, 0);
  gemm_wmma<<<gg, 256, 0, stream>>>(k16, wkT, bk, Kh, nullptr, 0);
  gemm_wmma<<<gg, 256, 0, stream>>>(vx16, wvT, bv, VhT, nullptr, 2);

  flash_attn<<<dim3(SEQ / FA_QB, BATCH * N_HEADS), 128, 0, stream>>>(Qh, Kh,
                                                                     VhT, AO);

  gemm_wmma<<<gg, 256, 0, stream>>>(AO, woT, bo, nullptr, (float*)d_out, 1);
}
